// SampleMessagePassingLayer_34565896798312
// MI455X (gfx1250) — matrix operand order repaired and checked by reading, not compiled, not run on hardware
//
#include <hip/hip_runtime.h>
#include <hip/hip_bf16.h>

#define N_NODES 50000
#define N_EDGES 800000
#define HID 128
#define INP 128
#define TDIM 64
#define CATM 448   // 2*HID + INP + TDIM

#define EK_WAVES   16
#define EK_THREADS (EK_WAVES * 32)
#define EK_EDGES   (EK_WAVES * 16)          // 256 edges per block
#define LDS_WM1_OFF 0u
#define LDS_WM2_OFF 114688u                  // 448*128*2
#define LDS_H_OFF   147456u                  // + 128*128*2
#define EK_LDS_BYTES (147456u + EK_WAVES * 16 * HID * 2)  // 212992

typedef _Float16 v16h __attribute__((ext_vector_type(16)));
typedef _Float16 v8h  __attribute__((ext_vector_type(8)));
typedef _Float16 v4h  __attribute__((ext_vector_type(4)));
typedef __fp16   v2hp __attribute__((ext_vector_type(2)));   // cvt_pkrtz result type
typedef float    v8f  __attribute__((ext_vector_type(8)));
typedef unsigned int v4u __attribute__((ext_vector_type(4)));
typedef int      v8i  __attribute__((ext_vector_type(8)));
typedef int      v4i  __attribute__((ext_vector_type(4)));

// ---------- WMMA helpers ----------

// A-matrix 16x32 f16 fragment from LDS row-major storage.
// lane: m = lane%16, hf = lane/16; halves 0..7 = K[k0+hf*8 ..+8),
// halves 8..15 = K[k0+16+hf*8 ..+8).
__device__ __forceinline__ v16h ldsA(const _Float16* rowbase, int stride,
                                     int m, int hf, int k0) {
    const _Float16* p = rowbase + m * stride + k0 + hf * 8;
    v8h lo = *(const v8h*)(p);
    v8h hi = *(const v8h*)(p + 16);
    return __builtin_shufflevector(lo, hi, 0,1,2,3,4,5,6,7,8,9,10,11,12,13,14,15);
}

// B-matrix 32x16 f16 fragment: weights transposed wT[n][k] (row length K).
// lane n = lane%16; halves 0..7 = K[k0+hf*8 ..+8), halves 8..15 = K[k0+16+hf*8 ..+8).
__device__ __forceinline__ v16h ldsB(const _Float16* wT, int K,
                                     int n, int hf, int k0) {
    const _Float16* p = wT + n * K + k0 + hf * 8;
    v8h lo = *(const v8h*)(p);
    v8h hi = *(const v8h*)(p + 16);
    return __builtin_shufflevector(lo, hi, 0,1,2,3,4,5,6,7,8,9,10,11,12,13,14,15);
}

__device__ __forceinline__ v8f wmma_f16(v16h a, v16h b, v8f c) {
    return __builtin_amdgcn_wmma_f32_16x16x32_f16(
        false, a, false, b, (short)0, c, false, false);
}

__device__ __forceinline__ float gelu_exact(float xv) {
    return 0.5f * xv * (1.0f + erff(xv * 0.70710678118654752f));
}

// Build A fragment directly from 8+8 consecutive f32 (global, L2-resident).
__device__ __forceinline__ v16h cvtA(const float* p) {
    float4 a0 = *(const float4*)(p);
    float4 a1 = *(const float4*)(p + 4);
    float4 b0 = *(const float4*)(p + 16);
    float4 b1 = *(const float4*)(p + 20);
    v16h r;
    v2hp t;
    t = __builtin_amdgcn_cvt_pkrtz(a0.x, a0.y); r[0]=t[0];  r[1]=t[1];
    t = __builtin_amdgcn_cvt_pkrtz(a0.z, a0.w); r[2]=t[0];  r[3]=t[1];
    t = __builtin_amdgcn_cvt_pkrtz(a1.x, a1.y); r[4]=t[0];  r[5]=t[1];
    t = __builtin_amdgcn_cvt_pkrtz(a1.z, a1.w); r[6]=t[0];  r[7]=t[1];
    t = __builtin_amdgcn_cvt_pkrtz(b0.x, b0.y); r[8]=t[0];  r[9]=t[1];
    t = __builtin_amdgcn_cvt_pkrtz(b0.z, b0.w); r[10]=t[0]; r[11]=t[1];
    t = __builtin_amdgcn_cvt_pkrtz(b1.x, b1.y); r[12]=t[0]; r[13]=t[1];
    t = __builtin_amdgcn_cvt_pkrtz(b1.z, b1.w); r[14]=t[0]; r[15]=t[1];
    return r;
}

__device__ __forceinline__ v16h cvtAdiff(const float* pa, const float* pb) {
    float4 a0 = *(const float4*)(pa),      c0 = *(const float4*)(pb);
    float4 a1 = *(const float4*)(pa + 4),  c1 = *(const float4*)(pb + 4);
    float4 b0 = *(const float4*)(pa + 16), d0 = *(const float4*)(pb + 16);
    float4 b1 = *(const float4*)(pa + 20), d1 = *(const float4*)(pb + 20);
    v16h r;
    v2hp t;
    t = __builtin_amdgcn_cvt_pkrtz(a0.x-c0.x, a0.y-c0.y); r[0]=t[0];  r[1]=t[1];
    t = __builtin_amdgcn_cvt_pkrtz(a0.z-c0.z, a0.w-c0.w); r[2]=t[0];  r[3]=t[1];
    t = __builtin_amdgcn_cvt_pkrtz(a1.x-c1.x, a1.y-c1.y); r[4]=t[0];  r[5]=t[1];
    t = __builtin_amdgcn_cvt_pkrtz(a1.z-c1.z, a1.w-c1.w); r[6]=t[0];  r[7]=t[1];
    t = __builtin_amdgcn_cvt_pkrtz(b0.x-d0.x, b0.y-d0.y); r[8]=t[0];  r[9]=t[1];
    t = __builtin_amdgcn_cvt_pkrtz(b0.z-d0.z, b0.w-d0.w); r[10]=t[0]; r[11]=t[1];
    t = __builtin_amdgcn_cvt_pkrtz(b1.x-d1.x, b1.y-d1.y); r[12]=t[0]; r[13]=t[1];
    t = __builtin_amdgcn_cvt_pkrtz(b1.z-d1.z, b1.w-d1.w); r[14]=t[0]; r[15]=t[1];
    return r;
}

__device__ __forceinline__ void st4h(_Float16* p, float a, float b, float c, float d) {
    v4h o; o[0]=(_Float16)a; o[1]=(_Float16)b; o[2]=(_Float16)c; o[3]=(_Float16)d;
    *(v4h*)p = o;
}

// ---------- TDM: 2-D f16 tile, global -> LDS (D# per ISA ch.8) ----------
__device__ __forceinline__ void tdm_load_2d_f16(unsigned lds_off, const _Float16* gptr,
                                                unsigned dim0, unsigned dim1) {
    unsigned long long ga = (unsigned long long)(uintptr_t)gptr;
    v4u g0;
    g0[0] = 1u;                                            // count=1, user desc
    g0[1] = lds_off;                                       // LDS byte address
    g0[2] = (unsigned)ga;                                  // global_addr[31:0]
    g0[3] = ((unsigned)(ga >> 32) & 0x01FFFFFFu) | 0x80000000u; // [56:32] | type=2
    unsigned long long s1 = (unsigned long long)dim0 * dim1;
    v8i g1;
    g1[0] = (int)(1u << 16);                               // data_size = 2 bytes
    g1[1] = (int)((dim0 & 0xFFFFu) << 16);                 // tensor_dim0[15:0]
    g1[2] = (int)((dim0 >> 16) | ((dim1 & 0xFFFFu) << 16));// dim0 hi | dim1 lo
    g1[3] = (int)((dim1 >> 16) | ((dim0 & 0xFFFFu) << 16));// dim1 hi | tile_dim0
    g1[4] = (int)(dim1 & 0xFFFFu);                         // tile_dim1 (tile_dim2=0)
    g1[5] = (int)dim0;                                     // tensor_dim0_stride lo
    g1[6] = (int)((unsigned)(s1 & 0xFFFFu) << 16);         // stride0 hi=0 | stride1 lo
    g1[7] = (int)(s1 >> 16);                               // stride1 hi
    v4i z4 = {0, 0, 0, 0};
    v8i z8 = {0, 0, 0, 0, 0, 0, 0, 0};
#if __has_builtin(__builtin_amdgcn_tensor_load_to_lds)
    __builtin_amdgcn_tensor_load_to_lds(g0, g1, z4, z4, z8, 0);
#endif
}

__device__ __forceinline__ void tdm_wait0() {
#if __has_builtin(__builtin_amdgcn_s_wait_tensorcnt)
    __builtin_amdgcn_s_wait_tensorcnt((short)0);
#else
    asm volatile("s_wait_tensorcnt 0x0" ::: "memory");
#endif
}

// ---------- prep: f32 weights [K][N] -> f16 transposed [N][K] ----------
__global__ void wconv_kernel(const float* __restrict__ in, _Float16* __restrict__ out,
                             int K, int Nn) {
    int idx = blockIdx.x * 256 + threadIdx.x;
    if (idx < K * Nn) {
        int n = idx / K;
        int k = idx - n * K;
        out[idx] = (_Float16)in[k * Nn + n];
    }
}

__global__ void zero_kernel(float* __restrict__ p, int n) {
    int idx = blockIdx.x * 256 + threadIdx.x;
    if (idx < n) p[idx] = 0.0f;
}

// ---------- edge kernel: msg MLP + weighted scatter-add ----------
// 512 threads = 16 waves; weights staged in LDS once per block via TDM;
// each wave then owns 16 edges independently (no further barriers).
__global__ __launch_bounds__(EK_THREADS) void edge_kernel(
    const float* __restrict__ s, const float* __restrict__ x,
    const int* __restrict__ eidx, const float* __restrict__ ew,
    const float* __restrict__ t,
    const _Float16* __restrict__ Wm1T, const float* __restrict__ bm1,
    const _Float16* __restrict__ Wm2T, const float* __restrict__ bm2,
    float* __restrict__ agg)
{
    extern __shared__ char smem[];
    _Float16* wm1  = (_Float16*)(smem + LDS_WM1_OFF);  // [128][448] f16
    _Float16* wm2  = (_Float16*)(smem + LDS_WM2_OFF);  // [128][128] f16
    _Float16* hbuf = (_Float16*)(smem + LDS_H_OFF);    // [16 waves][16][128] f16

    const int lane = threadIdx.x & 31;
    const int wave = threadIdx.x >> 5;
    const int nl   = lane & 15;
    const int hf   = lane >> 4;

    if (wave == 0) {
        if (lane == 0) {
            tdm_load_2d_f16(LDS_WM1_OFF, Wm1T, CATM, HID);
            tdm_load_2d_f16(LDS_WM2_OFF, Wm2T, HID, HID);
        }
        tdm_wait0();
    }
    __syncthreads();

    const int ebase = blockIdx.x * EK_EDGES + wave * 16;
    // this lane's A row: edge m = lane%16
    const int src = eidx[ebase + nl];
    const int dst = eidx[N_EDGES + ebase + nl];
    const float* srow = s + (size_t)src * HID;
    const float* drow = s + (size_t)dst * HID;
    const float* xs   = x + (size_t)src * INP;
    const float* xd   = x + (size_t)dst * INP;
    const float* trow = t + (size_t)dst * TDIM;

    // ---- stage 1: h = gelu(feat @ Wm1 + bm1), K = 448, A built from global ----
    v8f acc[8] = {};
    #pragma unroll
    for (int kc = 0; kc < CATM / 32; ++kc) {
        const int k0  = kc * 32;
        const int off = (kc & 3) * 32 + hf * 8;   // offset inside 128-wide segment
        v16h a;
        if      (kc < 4)  a = cvtA(srow + off);
        else if (kc < 8)  a = cvtA(drow + off);
        else if (kc < 12) a = cvtAdiff(xs + off, xd + off);
        else              a = cvtA(trow + (kc - 12) * 32 + hf * 8);
        #pragma unroll
        for (int nt = 0; nt < 8; ++nt) {
            v16h b = ldsB(wm1, CATM, nt*16 + nl, hf, k0);
            acc[nt] = wmma_f16(a, b, acc[nt]);
        }
    }

    // bias + exact-erf GELU, write h to LDS (C layout -> row-major f16)
    _Float16* hrow = hbuf + wave * (16 * HID);
    #pragma unroll
    for (int nt = 0; nt < 8; ++nt) {
        float bias = bm1[nt*16 + nl];
        #pragma unroll
        for (int r = 0; r < 8; ++r) {
            float g = gelu_exact(acc[nt][r] + bias);
            hrow[(hf*8 + r) * HID + nt*16 + nl] = (_Float16)g;
        }
    }

    // ---- stage 2: msg = h @ Wm2 + bm2, K = 128 ----
    v8f acc2[8] = {};
    #pragma unroll
    for (int kc = 0; kc < HID / 32; ++kc) {
        int k0 = kc * 32;
        v16h a = ldsA(hrow, HID, nl, hf, k0);
        #pragma unroll
        for (int nt = 0; nt < 8; ++nt) {
            v16h b = ldsB(wm2, HID, nt*16 + nl, hf, k0);
            acc2[nt] = wmma_f16(a, b, acc2[nt]);
        }
    }

    // ---- weighted scatter-add: agg[dst] += msg * ew ----
    float bias2[8];
    #pragma unroll
    for (int nt = 0; nt < 8; ++nt) bias2[nt] = bm2[nt*16 + nl];
    #pragma unroll
    for (int r = 0; r < 8; ++r) {
        int er = ebase + hf*8 + r;                 // C row -> edge id
        int d  = eidx[N_EDGES + er];
        float w = ew[er];
        float* dp = agg + (size_t)d * HID + nl;
        #pragma unroll
        for (int nt = 0; nt < 8; ++nt) {
            atomicAdd(dp + nt*16, (acc2[nt][r] + bias2[nt]) * w);
        }
    }
}

// ---------- node kernel: out = gelu([s|agg] @ Wu1 + bu1) @ Wu2 + bu2 ----------
__global__ __launch_bounds__(64) void node_kernel(
    const float* __restrict__ s, const float* __restrict__ agg,
    const _Float16* __restrict__ Wu1T, const float* __restrict__ bu1,
    const _Float16* __restrict__ Wu2T, const float* __restrict__ bu2,
    float* __restrict__ out)
{
    __shared__ _Float16 ubuf[32 * 2*HID];  // 16 KB
    __shared__ _Float16 h2buf[32 * HID];   // 8 KB

    const int lane = threadIdx.x & 31;
    const int wave = threadIdx.x >> 5;
    const int nl   = lane & 15;
    const int hf   = lane >> 4;
    const int nbase = blockIdx.x * 32 + wave * 16;

    _Float16* urow = ubuf + wave * 16 * (2*HID);
    _Float16* hrow = h2buf + wave * 16 * HID;

    for (int e = 0; e < 16; ++e) {
        int node = nbase + e;
        int cn = node < N_NODES ? node : (N_NODES - 1);
        _Float16* r = urow + e * (2*HID);
        const float* ps = s + (size_t)cn * HID + lane * 4;
        st4h(r + lane*4,       ps[0], ps[1], ps[2], ps[3]);
        const float* pa = agg + (size_t)cn * HID + lane * 4;
        st4h(r + HID + lane*4, pa[0], pa[1], pa[2], pa[3]);
    }

    // stage 1: K = 256
    v8f acc[8] = {};
    for (int kc = 0; kc < (2*HID) / 32; ++kc) {
        int k0 = kc * 32;
        v16h a = ldsA(urow, 2*HID, nl, hf, k0);
        #pragma unroll
        for (int nt = 0; nt < 8; ++nt) {
            v16h b = ldsB(Wu1T, 2*HID, nt*16 + nl, hf, k0);  // global (L2)
            acc[nt] = wmma_f16(a, b, acc[nt]);
        }
    }
    #pragma unroll
    for (int nt = 0; nt < 8; ++nt) {
        float bias = bu1[nt*16 + nl];
        #pragma unroll
        for (int r = 0; r < 8; ++r) {
            float g = gelu_exact(acc[nt][r] + bias);
            hrow[(hf*8 + r) * HID + nt*16 + nl] = (_Float16)g;
        }
    }

    // stage 2: K = 128
    v8f acc2[8] = {};
    for (int kc = 0; kc < HID / 32; ++kc) {
        int k0 = kc * 32;
        v16h a = ldsA(hrow, HID, nl, hf, k0);
        #pragma unroll
        for (int nt = 0; nt < 8; ++nt) {
            v16h b = ldsB(Wu2T, HID, nt*16 + nl, hf, k0);    // global (L2)
            acc2[nt] = wmma_f16(a, b, acc2[nt]);
        }
    }

    float bias2[8];
    #pragma unroll
    for (int nt = 0; nt < 8; ++nt) bias2[nt] = bu2[nt*16 + nl];
    #pragma unroll
    for (int r = 0; r < 8; ++r) {
        int node = nbase + hf*8 + r;
        if (node < N_NODES) {
            float* po = out + (size_t)node * HID + nl;
            #pragma unroll
            for (int nt = 0; nt < 8; ++nt)
                po[nt*16] = acc2[nt][r] + bias2[nt];
        }
    }
}

extern "C" void kernel_launch(void* const* d_in, const int* in_sizes, int n_in,
                              void* d_out, int out_size, void* d_ws, size_t ws_size,
                              hipStream_t stream)
{
    (void)in_sizes; (void)n_in; (void)out_size; (void)ws_size;
    const float* s    = (const float*)d_in[0];
    const float* x    = (const float*)d_in[1];
    const int*   eidx = (const int*)d_in[2];
    const float* ew   = (const float*)d_in[3];
    const float* t    = (const float*)d_in[4];
    const float* Wm1  = (const float*)d_in[5];
    const float* bm1  = (const float*)d_in[6];
    const float* Wm2  = (const float*)d_in[7];
    const float* bm2  = (const float*)d_in[8];
    const float* Wu1  = (const float*)d_in[9];
    const float* bu1  = (const float*)d_in[10];
    const float* Wu2  = (const float*)d_in[11];
    const float* bu2  = (const float*)d_in[12];
    float* out = (float*)d_out;

    char* ws = (char*)d_ws;
    _Float16* Wm1T = (_Float16*)(ws);             // 448*128 f16 = 114688 B
    _Float16* Wm2T = (_Float16*)(ws + 114688);    // 128*128 f16 =  32768 B
    _Float16* Wu1T = (_Float16*)(ws + 147456);    // 256*128 f16 =  65536 B
    _Float16* Wu2T = (_Float16*)(ws + 212992);    // 128*128 f16 =  32768 B
    float*    agg  = (float*)(ws + 245760);       // N*HID f32   = 25.6 MB

    // allow >64KB dynamic LDS on the edge kernel (WGP has 320 KB)
    (void)hipFuncSetAttribute(reinterpret_cast<const void*>(edge_kernel),
                              hipFuncAttributeMaxDynamicSharedMemorySize,
                              (int)EK_LDS_BYTES);

    wconv_kernel<<<(CATM*HID + 255)/256, 256, 0, stream>>>(Wm1, Wm1T, CATM, HID);
    wconv_kernel<<<(HID*HID + 255)/256, 256, 0, stream>>>(Wm2, Wm2T, HID, HID);
    wconv_kernel<<<(2*HID*HID + 255)/256, 256, 0, stream>>>(Wu1, Wu1T, 2*HID, HID);
    wconv_kernel<<<(HID*HID + 255)/256, 256, 0, stream>>>(Wu2, Wu2T, HID, HID);

    zero_kernel<<<(N_NODES*HID + 255)/256, 256, 0, stream>>>(agg, N_NODES*HID);

    edge_kernel<<<N_EDGES/EK_EDGES, EK_THREADS, EK_LDS_BYTES, stream>>>(
        s, x, eidx, ew, t, Wm1T, bm1, Wm2T, bm2, agg);
    node_kernel<<<(N_NODES + 31)/32, 64, 0, stream>>>(s, agg,
                                                      Wu1T, bu1, Wu2T, bu2, out);
}
